// MambaBlock_27212912788292
// MI455X (gfx1250) — matrix operand order repaired and checked by reading, not compiled, not run on hardware
//
#include <hip/hip_runtime.h>

// ---------------- problem constants ----------------
#define T_LEN 16384
#define DIMN  1024
#define SDIM  32
#define CHUNK 128
#define NCHUNK (T_LEN / CHUNK)   // 128

typedef __attribute__((ext_vector_type(16))) __bf16 v16bf;
typedef __attribute__((ext_vector_type(8)))  float  v8f;
typedef __attribute__((ext_vector_type(2)))  float  v2f;

__device__ __forceinline__ float fast_silu(float v) {
    // v * sigmoid(v) with hardware rcp (v_rcp_f32)
    return v * __builtin_amdgcn_rcpf(1.f + __expf(-v));
}

// =====================================================================
// Kernel 1: fused  u = silu(x @ W_in^T),  delta = clip(softplus(x.dp)),
//           a_t = exp(delta*A_bar),  b_t = (a_t-1)/|A_bar| * B * u
// One wave (32 lanes) per 16-row tile of T.  bf16 WMMA 16x16x32.
// =====================================================================
__global__ __launch_bounds__(256) void mamba_front(
    const float* __restrict__ x,       // [T, DIM]
    const float* __restrict__ W_in,    // [S, DIM]
    const float* __restrict__ A_log,   // [S]
    const float* __restrict__ B_mat,   // [S]
    const float* __restrict__ dp,      // [DIM]
    float* __restrict__ aArr,          // [T, S]
    float* __restrict__ bArr)          // [T, S]
{
    const int lane = threadIdx.x & 31;
    const int wave = threadIdx.x >> 5;
    const int tile = blockIdx.x * 8 + wave;     // 0 .. T/16-1
    const int t0   = tile * 16;
    const int half = lane >> 4;                 // 0: lanes 0-15, 1: lanes 16-31
    const int m    = lane & 15;

    const float* __restrict__ xrow = x + (size_t)(t0 + m) * DIMN;
    // B-fragment base: lane = output state column, half-lanes split the K window
    const float* __restrict__ w0 = W_in + (size_t)m        * DIMN + half * 8;
    const float* __restrict__ w1 = W_in + (size_t)(m + 16) * DIMN + half * 8;

    v8f acc0 = {};   // states 0..15
    v8f acc1 = {};   // states 16..31
    // 4 independent delta accumulators: break the serial v_fmac chain
    float dA0 = 0.f, dA1 = 0.f, dA2 = 0.f, dA3 = 0.f;

    for (int ko = 0; ko < DIMN; ko += 32) {
        if (ko + 64 < DIMN) {
            __builtin_prefetch(xrow + ko + 64, 0, 1);   // global_prefetch_b8
        }
        v16bf afrag, bfrag0, bfrag1;
        // A fragment (x tile, 16x32 bf16): ISA layout — lane row M = lane&15,
        // VGPR v (pair e=2v,2v+1): K = half*8 + (v<4 ? 2v : 16+2(v-4)) + {0,1}
        #pragma unroll
        for (int v = 0; v < 8; ++v) {
            const int kb = ko + half * 8 + (v < 4 ? 2 * v : 16 + 2 * (v - 4));
            const float x0 = xrow[kb];
            const float x1 = xrow[kb + 1];
            if ((v & 1) == 0) {
                dA0 = fmaf(x0, dp[kb],     dA0);
                dA1 = fmaf(x1, dp[kb + 1], dA1);
            } else {
                dA2 = fmaf(x0, dp[kb],     dA2);
                dA3 = fmaf(x1, dp[kb + 1], dA3);
            }
            afrag[2 * v]     = (__bf16)x0;
            afrag[2 * v + 1] = (__bf16)x1;
        }
        // B fragment (W_in^T, 32x16 bf16): lane col N, K = half*8 + (e<8 ? e : 16+(e-8))
        #pragma unroll
        for (int e = 0; e < 16; ++e) {
            bfrag0[e] = (__bf16)w0[ko + (e < 8 ? e : e + 8)];
            bfrag1[e] = (__bf16)w1[ko + (e < 8 ? e : e + 8)];
        }
        acc0 = __builtin_amdgcn_wmma_f32_16x16x32_bf16(
                 false, afrag, false, bfrag0, (short)0, acc0, false, false);
        acc1 = __builtin_amdgcn_wmma_f32_16x16x32_bf16(
                 false, afrag, false, bfrag1, (short)0, acc1, false, false);
    }

    float dAcc = (dA0 + dA1) + (dA2 + dA3);
    // lanes L and L+16 jointly covered row m's K range: combine
    dAcc += __shfl_xor(dAcc, 16, 32);   // full dot(x[t0+m], dp) on all lanes

    // per-lane state constants (column of C = lane&15 / +16)
    const float ab0  = __expf(A_log[m]);
    const float ab1  = __expf(A_log[m + 16]);
    const float rden0 = __builtin_amdgcn_rcpf(fmaxf(fabsf(ab0), 1e-8f));
    const float rden1 = __builtin_amdgcn_rcpf(fmaxf(fabsf(ab1), 1e-8f));
    const float bm0  = B_mat[m];
    const float bm1  = B_mat[m + 16];

    #pragma unroll
    for (int r = 0; r < 8; ++r) {
        // C layout: VGPR r holds row M = r + 8*half, col N = lane&15
        const int rowInTile = half * 8 + r;
        const int t = t0 + rowInTile;
        // delta for this row lives in lane 'rowInTile' (<16, holds full sum)
        float d  = __shfl(dAcc, rowInTile, 32);
        // fast softplus: argument is O(1) here (|dp| <= 0.01, dot of 1024)
        float sp = (d > 20.f) ? d : __logf(1.f + __expf(d));
        float dt = fminf(fmaxf(sp, 0.001f), 0.1f);

        float u0 = fast_silu(acc0[r]);
        float u1 = fast_silu(acc1[r]);

        float a0 = __expf(dt * ab0);
        float a1 = __expf(dt * ab1);
        float b0 = (a0 - 1.f) * rden0 * bm0 * u0;
        float b1 = (a1 - 1.f) * rden1 * bm1 * u1;

        aArr[(size_t)t * SDIM + m]      = a0;
        aArr[(size_t)t * SDIM + m + 16] = a1;
        bArr[(size_t)t * SDIM + m]      = b0;
        bArr[(size_t)t * SDIM + m + 16] = b1;
    }
}

// =====================================================================
// Kernel 2: per-chunk scan aggregates.  wave = chunk, lane = state.
//   P_c = prod a_t,  H_c = scan(chunk, h0=0) final value
// =====================================================================
__global__ __launch_bounds__(256) void mamba_chunk(
    const float* __restrict__ aArr, const float* __restrict__ bArr,
    float* __restrict__ chunkP, float* __restrict__ chunkH)
{
    const int lane = threadIdx.x & 31;
    const int c = blockIdx.x * 8 + (threadIdx.x >> 5);
    if (c >= NCHUNK) return;
    float p = 1.f, h = 0.f;
    const int base = c * CHUNK;
    for (int i = 0; i < CHUNK; ++i) {
        const float a = aArr[(size_t)(base + i) * SDIM + lane];
        const float b = bArr[(size_t)(base + i) * SDIM + lane];
        p *= a;
        h = fmaf(a, h, b);
    }
    chunkP[c * SDIM + lane] = p;
    chunkH[c * SDIM + lane] = h;
}

// =====================================================================
// Kernel 3: exclusive scan of chunk aggregates (1 wave, lane = state)
// =====================================================================
__global__ void mamba_carry(
    const float* __restrict__ chunkP, const float* __restrict__ chunkH,
    float* __restrict__ carry)
{
    const int s = threadIdx.x;
    float h = 0.f;
    for (int c = 0; c < NCHUNK; ++c) {
        carry[c * SDIM + s] = h;
        h = fmaf(chunkP[c * SDIM + s], h, chunkH[c * SDIM + s]);
    }
}

// =====================================================================
// Kernel 4: re-scan chunk with carry into LDS, then y = hs @ W_out^T
// via f32 WMMA 16x16x4 (K=32 -> 8 wmma per 16x16 output tile).
// Block = 8 waves per chunk; wave w owns row-tile w, sweeps 64 col-tiles.
// =====================================================================
__global__ __launch_bounds__(256) void mamba_out(
    const float* __restrict__ aArr, const float* __restrict__ bArr,
    const float* __restrict__ carry, const float* __restrict__ W_out, // [DIM, S]
    float* __restrict__ y)                                            // [T, DIM]
{
    __shared__ float hs[CHUNK][SDIM + 1];   // +1 pad: conflict-free strided reads

    const int lane  = threadIdx.x & 31;
    const int wave  = threadIdx.x >> 5;
    const int c     = blockIdx.x;
    const int tbase = c * CHUNK;

    if (wave == 0) {                         // sequential local scan, lane = state
        float h = carry[c * SDIM + lane];
        for (int i = 0; i < CHUNK; ++i) {
            const float a = aArr[(size_t)(tbase + i) * SDIM + lane];
            const float b = bArr[(size_t)(tbase + i) * SDIM + lane];
            h = fmaf(a, h, b);
            hs[i][lane] = h;
        }
    }
    __syncthreads();

    const int half = lane >> 4;
    const int m    = lane & 15;
    const int r0   = wave * 16;              // this wave's 16 rows within chunk

    // Preload all A fragments for K = 0..31 (8 windows of 4).
    // f32 16x16x4 A layout: lane row M = lane&15; VGPR0/1 = K = 2*half + {0,1}
    v2f afr[8];
    #pragma unroll
    for (int w = 0; w < 8; ++w) {
        const int k0 = w * 4 + half * 2;
        afr[w][0] = hs[r0 + m][k0];
        afr[w][1] = hs[r0 + m][k0 + 1];
    }

    for (int db = 0; db < DIMN; db += 16) {
        v8f acc = {};
        const float* __restrict__ wrow = W_out + (size_t)(db + m) * SDIM;
        #pragma unroll
        for (int w = 0; w < 8; ++w) {
            const int k0 = w * 4 + half * 2;
            v2f bfr;                         // B[k][n] = W_out[n][k]
            bfr[0] = wrow[k0];
            bfr[1] = wrow[k0 + 1];
            acc = __builtin_amdgcn_wmma_f32_16x16x4_f32(
                    false, afr[w], false, bfr, (short)0, acc, false, false);
        }
        #pragma unroll
        for (int vr = 0; vr < 8; ++vr) {     // C: VGPR vr -> row vr + 8*half
            const int t = tbase + r0 + vr + half * 8;
            y[(size_t)t * DIMN + db + m] = acc[vr];
        }
    }
}

// =====================================================================
extern "C" void kernel_launch(void* const* d_in, const int* in_sizes, int n_in,
                              void* d_out, int out_size, void* d_ws, size_t ws_size,
                              hipStream_t stream) {
    const float* x     = (const float*)d_in[0];   // [T, DIM]
    const float* W_in  = (const float*)d_in[1];   // [S, DIM]
    const float* W_out = (const float*)d_in[2];   // [DIM, S]
    const float* A_log = (const float*)d_in[3];   // [S]
    const float* B_mat = (const float*)d_in[4];   // [S]
    const float* dp    = (const float*)d_in[5];   // [DIM]
    float* y = (float*)d_out;                     // [T, DIM]

    // workspace layout (floats)
    float* ws     = (float*)d_ws;
    float* aArr   = ws;                               // T*S
    float* bArr   = aArr   + (size_t)T_LEN * SDIM;    // T*S
    float* chunkP = bArr   + (size_t)T_LEN * SDIM;    // NC*S
    float* chunkH = chunkP + (size_t)NCHUNK * SDIM;   // NC*S
    float* carry  = chunkH + (size_t)NCHUNK * SDIM;   // NC*S

    mamba_front<<<T_LEN / 16 / 8, 256, 0, stream>>>(x, W_in, A_log, B_mat, dp,
                                                    aArr, bArr);
    mamba_chunk<<<NCHUNK / 8, 256, 0, stream>>>(aArr, bArr, chunkP, chunkH);
    mamba_carry<<<1, SDIM, 0, stream>>>(chunkP, chunkH, carry);
    mamba_out<<<NCHUNK, 256, 0, stream>>>(aArr, bArr, carry, W_out, y);
}
